// Attention_13666585936584
// MI455X (gfx1250) — matrix operand order repaired and checked by reading, not compiled, not run on hardware
//
#include <hip/hip_runtime.h>
#include <hip/hip_bf16.h>
#include <stdint.h>

// ---------------------------------------------------------------------------
// MI455X (gfx1250) multi-head attention, bf16 WMMA (16x16x32), f32 accumulate.
//   x[2,2048,1024] -> qkv GEMM -> flash attention per head -> proj GEMM
// V is stored pre-transposed [B,H,D,N] so attention stages K and V^T into LDS
// with GLOBAL_LOAD_ASYNC_TO_LDS_B128 (ASYNCcnt) -- no VGPR round trip.
// f32->bf16 staging uses V_CVT_PK_BF16_F32 (1 VALU op / 2 elements).
// ---------------------------------------------------------------------------

#define DIMC   1024
#define NHEADS 16
#define HD     64
#define SEQ    2048
#define BATCH  2

typedef __attribute__((ext_vector_type(16))) __bf16 bf16x16;
typedef __attribute__((ext_vector_type(8)))  __bf16 bf16x8;
typedef __attribute__((ext_vector_type(8)))  float  f32x8;

#define WMMA_BF16(a, b, c) \
    __builtin_amdgcn_wmma_f32_16x16x32_bf16(false, (a), false, (b), (short)0, (c), false, false)

__device__ __forceinline__ f32x8 zero8() {
    f32x8 z = {0.f, 0.f, 0.f, 0.f, 0.f, 0.f, 0.f, 0.f};
    return z;
}

__device__ __forceinline__ unsigned short f32_to_bf16(float f) {
    unsigned int u = __builtin_bit_cast(unsigned int, f);
    unsigned int r = u + 0x7FFFu + ((u >> 16) & 1u);   // round-to-nearest-even
    return (unsigned short)(r >> 16);
}

// Two f32 -> packed bf16 pair in one VALU op (V_CVT_PK_BF16_F32).
__device__ __forceinline__ unsigned pack_bf16(float a, float b) {
    unsigned r;
    asm("v_cvt_pk_bf16_f32 %0, %1, %2" : "=v"(r) : "v"(a), "v"(b));
    return r;
}
__device__ __forceinline__ void store_bf16x2(unsigned short* d, float a, float b) {
    *reinterpret_cast<unsigned*>(d) = pack_bf16(a, b);
}

// Async 16-byte global -> LDS copy (per lane), tracked by ASYNCcnt.
__device__ __forceinline__ void async_copy16(unsigned short* lds_dst,
                                             const unsigned short* gsrc) {
    unsigned lds_off = (unsigned)(uintptr_t)lds_dst;   // low 32 bits = LDS offset
    asm volatile("global_load_async_to_lds_b128 %0, %1, off"
                 :: "v"(lds_off), "v"(gsrc) : "memory");
}

__device__ __forceinline__ void wait_async0() {
#if __has_builtin(__builtin_amdgcn_s_wait_asynccnt)
    __builtin_amdgcn_s_wait_asynccnt(0);
#else
    asm volatile("s_wait_asynccnt 0" ::: "memory");
#endif
}

__device__ __forceinline__ bf16x8 ld8(const unsigned short* p) {
    return *reinterpret_cast<const bf16x8*>(p);
}
__device__ __forceinline__ bf16x16 cat16(bf16x8 lo, bf16x8 hi) {
    return __builtin_shufflevector(lo, hi, 0,1,2,3,4,5,6,7,8,9,10,11,12,13,14,15);
}

// A fragment (16x32 bf16): lane = M (mod 16); lanes<16 hold K {0..7,16..23},
// lanes>=16 hold K {8..15,24..31}. Tile is row-major [row][k] in LDS.
__device__ __forceinline__ bf16x16 load_a_frag(const unsigned short* t, int stride,
                                               int row0, int k0, int lane) {
    const unsigned short* p = t + (row0 + (lane & 15)) * stride + k0 + ((lane < 16) ? 0 : 8);
    return cat16(ld8(p), ld8(p + 16));
}

// B fragment (32x16 bf16), B[k][n] = Wtile[n][k]: lane = N (mod 16);
// lanes<16 hold K {0..7,16..23}, lanes>=16 hold K {8..15,24..31}.
__device__ __forceinline__ bf16x16 load_b_frag(const unsigned short* t, int stride,
                                               int n0, int k0, int lane) {
    const unsigned short* p = t + (n0 + (lane & 15)) * stride + k0 + ((lane < 16) ? 0 : 8);
    return cat16(ld8(p), ld8(p + 16));
}

// ---------------------------------------------------------------------------
// Kernel 1: qkv = x @ qkv_w^T + qkv_b
//   Q scaled by 1/sqrt(64), layout [B,H,N,D]; K layout [B,H,N,D];
//   V stored TRANSPOSED as [B,H,D,N] for async staging in attention.
// ---------------------------------------------------------------------------
__global__ __launch_bounds__(256) void qkv_kernel(
    const float* __restrict__ x, const float* __restrict__ w,
    const float* __restrict__ bias,
    unsigned short* __restrict__ Qb, unsigned short* __restrict__ Kb,
    unsigned short* __restrict__ Vb)
{
    const int M0 = blockIdx.y * 128;
    const int R0 = blockIdx.x * 128;
    const int tid = threadIdx.x, lane = tid & 31, wave = tid >> 5;
    const int wr = wave & 3, wc = wave >> 2;

    __shared__ __align__(16) unsigned short As[128 * 40];
    __shared__ __align__(16) unsigned short Bs[128 * 40];

    f32x8 acc[2][4];
#pragma unroll
    for (int i = 0; i < 2; i++)
#pragma unroll
        for (int j = 0; j < 4; j++) acc[i][j] = zero8();

    for (int k0 = 0; k0 < DIMC; k0 += 32) {
        if (k0 + 32 < DIMC) {
            __builtin_prefetch(x + (size_t)(M0 + (tid >> 1)) * DIMC + k0 + 32, 0, 3);
            __builtin_prefetch(w + (size_t)(R0 + (tid >> 1)) * DIMC + k0 + 32, 0, 3);
        }
#pragma unroll
        for (int c = tid; c < 128 * 8; c += 256) {      // stage x tile -> bf16
            int row = c >> 3, cc = (c & 7) * 4;
            float4 v = *reinterpret_cast<const float4*>(x + (size_t)(M0 + row) * DIMC + k0 + cc);
            unsigned short* d = As + row * 40 + cc;
            store_bf16x2(d, v.x, v.y);
            store_bf16x2(d + 2, v.z, v.w);
        }
#pragma unroll
        for (int c = tid; c < 128 * 8; c += 256) {      // stage w tile -> bf16
            int row = c >> 3, cc = (c & 7) * 4;
            float4 v = *reinterpret_cast<const float4*>(w + (size_t)(R0 + row) * DIMC + k0 + cc);
            unsigned short* d = Bs + row * 40 + cc;
            store_bf16x2(d, v.x, v.y);
            store_bf16x2(d + 2, v.z, v.w);
        }
        __syncthreads();
        bf16x16 af[2], bf[4];
#pragma unroll
        for (int mi = 0; mi < 2; mi++) af[mi] = load_a_frag(As, 40, wr * 32 + mi * 16, 0, lane);
#pragma unroll
        for (int ni = 0; ni < 4; ni++) bf[ni] = load_b_frag(Bs, 40, wc * 64 + ni * 16, 0, lane);
#pragma unroll
        for (int mi = 0; mi < 2; mi++)
#pragma unroll
            for (int ni = 0; ni < 4; ni++)
                acc[mi][ni] = WMMA_BF16(af[mi], bf[ni], acc[mi][ni]);
        __syncthreads();
    }

    const int rg = (lane < 16) ? 0 : 8;
#pragma unroll
    for (int mi = 0; mi < 2; mi++)
#pragma unroll
        for (int ni = 0; ni < 4; ni++) {
            int r = R0 + wc * 64 + ni * 16 + (lane & 15);
            float bv = bias[r];
            int sel = r >> 10, h = (r & 1023) >> 6, d = r & 63;
            if (sel == 2) {                               // V: transposed [B,H,D,N]
#pragma unroll
                for (int i = 0; i < 8; i++) {
                    int m = M0 + wr * 32 + mi * 16 + i + rg;
                    int b = m >> 11, n = m & 2047;
                    Vb[(((size_t)(b * NHEADS + h)) * HD + d) * SEQ + n] =
                        f32_to_bf16(acc[mi][ni][i] + bv);
                }
            } else {                                      // Q (scaled) / K: [B,H,N,D]
                unsigned short* dst = (sel == 0) ? Qb : Kb;
                float mul = (sel == 0) ? 0.125f : 1.0f;   // fold softmax scale into Q
#pragma unroll
                for (int i = 0; i < 8; i++) {
                    int m = M0 + wr * 32 + mi * 16 + i + rg;
                    int b = m >> 11, n = m & 2047;
                    dst[(((size_t)(b * NHEADS + h)) * SEQ + n) * HD + d] =
                        f32_to_bf16((acc[mi][ni][i] + bv) * mul);
                }
            }
        }
}

// ---------------------------------------------------------------------------
// Kernel 2: flash attention, one WG per (b,h, 128 Q rows), KV blocks of 64.
// K and V^T tiles staged with async global->LDS copies.
// ---------------------------------------------------------------------------
__global__ __launch_bounds__(256) void attn_kernel(
    const unsigned short* __restrict__ Qb, const unsigned short* __restrict__ Kb,
    const unsigned short* __restrict__ Vb, unsigned short* __restrict__ Ob)
{
    const int bh = blockIdx.y;                 // b*16+h
    const int q0 = blockIdx.x * 128;
    const int b = bh >> 4, h = bh & 15;
    const int tid = threadIdx.x, lane = tid & 31, wave = tid >> 5;

    __shared__ __align__(16) unsigned short Qs[128 * 72];
    __shared__ __align__(16) unsigned short Ks[64 * 72];
    __shared__ __align__(16) unsigned short Vts[64 * 72];      // [d][j]
    __shared__ __align__(16) unsigned short Ps[8 * 16 * 72];   // per-wave P strips

    const unsigned short* Qg = Qb + ((size_t)bh * SEQ + q0) * HD;
    const unsigned short* Kg = Kb + (size_t)bh * SEQ * HD;
    const unsigned short* Vg = Vb + (size_t)bh * HD * SEQ;     // transposed [d][n]

#pragma unroll
    for (int c = tid; c < 128 * 8; c += 256) {                 // Q block (async)
        int row = c >> 3, cc = (c & 7) * 8;
        async_copy16(Qs + row * 72 + cc, Qg + (size_t)row * HD + cc);
    }
    wait_async0();
    __syncthreads();

    bf16x16 qf[2];                              // Q fragments live in registers
#pragma unroll
    for (int kk = 0; kk < 2; kk++) qf[kk] = load_a_frag(Qs, 72, wave * 16, kk * 32, lane);

    f32x8 accO[4];
#pragma unroll
    for (int i = 0; i < 4; i++) accO[i] = zero8();
    float mrow[8], lrow[8];
#pragma unroll
    for (int i = 0; i < 8; i++) { mrow[i] = -3.0e38f; lrow[i] = 0.0f; }

    for (int j0 = 0; j0 < SEQ; j0 += 64) {
        __syncthreads();                        // previous block fully consumed
#pragma unroll
        for (int c = tid; c < 64 * 8; c += 256) {              // K block (async)
            int row = c >> 3, cc = (c & 7) * 8;
            async_copy16(Ks + row * 72 + cc, Kg + (size_t)(j0 + row) * HD + cc);
        }
#pragma unroll
        for (int c = tid; c < 64 * 8; c += 256) {              // V^T block (async)
            int row = c >> 3, cc = (c & 7) * 8;                // row = d, cc = j off
            async_copy16(Vts + row * 72 + cc, Vg + (size_t)row * SEQ + j0 + cc);
        }
        wait_async0();
        __syncthreads();

        f32x8 s[4];                                         // S = Q K^T (16x64)
#pragma unroll
        for (int nt = 0; nt < 4; nt++) {
            s[nt] = zero8();
#pragma unroll
            for (int kk = 0; kk < 2; kk++) {
                bf16x16 bk = load_b_frag(Ks, 72, nt * 16, kk * 32, lane);
                s[nt] = WMMA_BF16(qf[kk], bk, s[nt]);
            }
        }

        float mnew[8], psum[8];
#pragma unroll
        for (int i = 0; i < 8; i++) {                       // row max (16-lane groups)
            float mx = fmaxf(fmaxf(s[0][i], s[1][i]), fmaxf(s[2][i], s[3][i]));
#pragma unroll
            for (int d = 1; d < 16; d <<= 1) mx = fmaxf(mx, __shfl_xor(mx, d, 32));
            mnew[i] = fmaxf(mrow[i], mx);
        }
#pragma unroll
        for (int i = 0; i < 8; i++) {
            float corr = __expf(mrow[i] - mnew[i]);
            lrow[i] *= corr;
#pragma unroll
            for (int dt = 0; dt < 4; dt++) accO[dt][i] *= corr;
            float p0 = __expf(s[0][i] - mnew[i]), p1 = __expf(s[1][i] - mnew[i]);
            float p2 = __expf(s[2][i] - mnew[i]), p3 = __expf(s[3][i] - mnew[i]);
            psum[i] = (p0 + p1) + (p2 + p3);
            unsigned short* pr = Ps + (wave * 16 + i + ((lane < 16) ? 0 : 8)) * 72 + (lane & 15);
            pr[0]  = f32_to_bf16(p0);  pr[16] = f32_to_bf16(p1);
            pr[32] = f32_to_bf16(p2);  pr[48] = f32_to_bf16(p3);
            mrow[i] = mnew[i];
        }
#pragma unroll
        for (int i = 0; i < 8; i++) {                       // row sum
            float sm = psum[i];
#pragma unroll
            for (int d = 1; d < 16; d <<= 1) sm += __shfl_xor(sm, d, 32);
            lrow[i] += sm;
        }
#pragma unroll
        for (int dt = 0; dt < 4; dt++)                      // O += P @ V
#pragma unroll
            for (int kk = 0; kk < 2; kk++) {
                bf16x16 ap = load_a_frag(Ps + wave * 16 * 72, 72, 0, kk * 32, lane);
                bf16x16 bv = load_b_frag(Vts, 72, dt * 16, kk * 32, lane);
                accO[dt] = WMMA_BF16(ap, bv, accO[dt]);
            }
    }

#pragma unroll
    for (int dt = 0; dt < 4; dt++)
#pragma unroll
        for (int i = 0; i < 8; i++) {
            int n = q0 + wave * 16 + i + ((lane < 16) ? 0 : 8);
            int d = dt * 16 + (lane & 15);
            Ob[((size_t)(b * SEQ + n)) * DIMC + h * HD + d] =
                f32_to_bf16(accO[dt][i] / lrow[i]);
        }
}

// ---------------------------------------------------------------------------
// Kernel 3: out = O @ proj_w^T + proj_b  (fp32 output); A staged async.
// ---------------------------------------------------------------------------
__global__ __launch_bounds__(256) void proj_kernel(
    const unsigned short* __restrict__ A, const float* __restrict__ w,
    const float* __restrict__ bias, float* __restrict__ out)
{
    const int M0 = blockIdx.y * 128, N0 = blockIdx.x * 128;
    const int tid = threadIdx.x, lane = tid & 31, wave = tid >> 5;
    const int wr = wave & 3, wc = wave >> 2;

    __shared__ __align__(16) unsigned short As[128 * 40];
    __shared__ __align__(16) unsigned short Bs[128 * 40];

    f32x8 acc[2][4];
#pragma unroll
    for (int i = 0; i < 2; i++)
#pragma unroll
        for (int j = 0; j < 4; j++) acc[i][j] = zero8();

    for (int k0 = 0; k0 < DIMC; k0 += 32) {
        if (k0 + 32 < DIMC) {
            __builtin_prefetch(A + (size_t)(M0 + (tid >> 1)) * DIMC + k0 + 32, 0, 3);
            __builtin_prefetch(w + (size_t)(N0 + (tid >> 1)) * DIMC + k0 + 32, 0, 3);
        }
#pragma unroll
        for (int c = tid; c < 128 * 4; c += 256) {          // A (bf16) async copy
            int row = c >> 2, cc = (c & 3) * 8;
            async_copy16(As + row * 40 + cc, A + (size_t)(M0 + row) * DIMC + k0 + cc);
        }
#pragma unroll
        for (int c = tid; c < 128 * 8; c += 256) {          // w f32 -> bf16
            int row = c >> 3, cc = (c & 7) * 4;
            float4 v = *reinterpret_cast<const float4*>(w + (size_t)(N0 + row) * DIMC + k0 + cc);
            unsigned short* d = Bs + row * 40 + cc;
            store_bf16x2(d, v.x, v.y);
            store_bf16x2(d + 2, v.z, v.w);
        }
        wait_async0();
        __syncthreads();
        bf16x16 af[2], bf[4];
#pragma unroll
        for (int mi = 0; mi < 2; mi++) af[mi] = load_a_frag(As, 40, wr * 32 + mi * 16, 0, lane);
#pragma unroll
        for (int ni = 0; ni < 4; ni++) bf[ni] = load_b_frag(Bs, 40, wc * 64 + ni * 16, 0, lane);
#pragma unroll
        for (int mi = 0; mi < 2; mi++)
#pragma unroll
            for (int ni = 0; ni < 4; ni++)
                acc[mi][ni] = WMMA_BF16(af[mi], bf[ni], acc[mi][ni]);
        __syncthreads();
    }

#pragma unroll
    for (int mi = 0; mi < 2; mi++)
#pragma unroll
        for (int ni = 0; ni < 4; ni++) {
            int n = N0 + wc * 64 + ni * 16 + (lane & 15);
            float bv = bias[n];
#pragma unroll
            for (int i = 0; i < 8; i++) {
                int m = M0 + wr * 32 + mi * 16 + i + ((lane < 16) ? 0 : 8);
                out[(size_t)m * DIMC + n] = acc[mi][ni][i] + bv;
            }
        }
}

// ---------------------------------------------------------------------------
extern "C" void kernel_launch(void* const* d_in, const int* in_sizes, int n_in,
                              void* d_out, int out_size, void* d_ws, size_t ws_size,
                              hipStream_t stream) {
    (void)in_sizes; (void)n_in; (void)out_size; (void)ws_size;
    const float* x      = (const float*)d_in[0];
    const float* qkv_w  = (const float*)d_in[1];
    const float* qkv_b  = (const float*)d_in[2];
    const float* proj_w = (const float*)d_in[3];
    const float* proj_b = (const float*)d_in[4];
    float* out = (float*)d_out;

    const size_t perBuf = (size_t)BATCH * NHEADS * SEQ * HD;   // 4M bf16 elems
    unsigned short* Qb = (unsigned short*)d_ws;
    unsigned short* Kb = Qb + perBuf;
    unsigned short* Vb = Kb + perBuf;
    unsigned short* Ob = Vb + perBuf;

    qkv_kernel <<<dim3(24, 32), 256, 0, stream>>>(x, qkv_w, qkv_b, Qb, Kb, Vb);
    attn_kernel<<<dim3(16, 32), 256, 0, stream>>>(Qb, Kb, Vb, Ob);
    proj_kernel<<<dim3(8, 32),  256, 0, stream>>>(Ob, proj_w, proj_b, out);
}
